// EALSTM_12189117186576
// MI455X (gfx1250) — compile-verified, matrix operand order checked by reading, not run on hardware
//
#include <hip/hip_runtime.h>

// ---- problem constants ----
#define B_    2048
#define T_    128
#define D_    32
#define H_    256
#define S_    39
#define PM_   12

#define BTILE   16                 // batch rows per workgroup
#define NWAVES  16                 // wave w owns H-coltile w for all 3 gates
#define NTHREADS (NWAVES * 32)
#define LDH   (H_ + 8)             // padded bf16 stride for s_h
#define LDX   (D_ + 8)             // padded bf16 stride for s_x
#define ACCP  (H_ + 4)             // padded f32 stride for s_acc
#define FRAG  512                  // bf16 elems per pre-swizzled B fragment (32 lanes x 16)
#define NFRAG 11                   // per-wave LDS frags: k=0..7 -> Wo_h, 8/9/10 -> Wf_i/Wg_i/Wo_i

typedef __attribute__((ext_vector_type(16))) __bf16 v16bf;
typedef __attribute__((ext_vector_type(8)))  __bf16 v8bf;
typedef __attribute__((ext_vector_type(8)))  float  v8f;

struct two_v8bf { v8bf lo, hi; };

__device__ __forceinline__ v8f wmma_bf16(v16bf a, v16bf b, v8f c) {
    // D = A(16x32 bf16) * B(32x16 bf16) + C(16x16 f32)
    return __builtin_amdgcn_wmma_f32_16x16x32_bf16(
        /*neg_a=*/false, a, /*neg_b=*/false, b,
        /*c_mod=*/(short)0, c, /*reuse_a=*/false, /*reuse_b=*/false);
}

// A-fragment (16x32, bf16) from LDS, row-major with stride `lds` (bf16 units).
// ISA 7.12.2: lanes 0-15 -> M=lane, K={0..7,16..23}; lanes 16-31 -> M=lane-16, K={8..15,24..31}
__device__ __forceinline__ v16bf load_afrag(const __bf16* base, int lds, int ko, int lane) {
    int m  = lane & 15;
    int kb = (lane < 16) ? 0 : 8;
    const __bf16* p = base + m * lds + ko + kb;
    two_v8bf t;
    t.lo = *(const v8bf*)(p);        // K = kb .. kb+7
    t.hi = *(const v8bf*)(p + 16);   // K = kb+16 .. kb+23
    return __builtin_bit_cast(v16bf, t);
}

// B-fragment (32x16, bf16) of W^T where W is row-major [nrows x ldw] f32.
// B[k][n] = W[ncol0 + n][ko + k].  Mirror of the A layout (N in place of M).
__device__ __forceinline__ v16bf load_bfrag(const float* __restrict__ W, int ldw,
                                            int ncol0, int ko, int lane) {
    int n  = ncol0 + (lane & 15);
    int kb = (lane < 16) ? 0 : 8;
    const float* p = W + (size_t)n * ldw + ko + kb;
    v16bf r;
#pragma unroll
    for (int j = 0; j < 8; ++j) r[j]     = (__bf16)p[j];
#pragma unroll
    for (int j = 0; j < 8; ++j) r[8 + j] = (__bf16)p[16 + j];
    return r;
}

// Pre-swizzled B-fragment in LDS: per frag, all lanes' lo chunks contiguous
// (16B per lane), then all hi chunks.
__device__ __forceinline__ v16bf load_frag(const __bf16* sw, int idx, int lane) {
    const __bf16* p = sw + (size_t)idx * FRAG + lane * 8;
    two_v8bf t;
    t.lo = *(const v8bf*)(p);
    t.hi = *(const v8bf*)(p + 256);
    return __builtin_bit_cast(v16bf, t);
}

__device__ __forceinline__ void store_frag(__bf16* sw, int idx, int lane, v16bf f) {
    two_v8bf t = __builtin_bit_cast(two_v8bf, f);
    __bf16* p = sw + (size_t)idx * FRAG + lane * 8;
    *(v8bf*)(p)       = t.lo;
    *(v8bf*)(p + 256) = t.hi;
}

__device__ __forceinline__ float fsig(float v)  { return 1.f / (1.f + __expf(-v)); }
__device__ __forceinline__ float ftanh(float v) { float e = __expf(2.f * v); return 1.f - 2.f / (e + 1.f); }

// Fold dense head: weff[0:268] = Wd1 @ Wd0 ; weff[268] = Wd1 @ bd0 + bd1
// (no nonlinearity between layers in the reference)
__global__ void head_fold_kernel(const float* __restrict__ Wd0, const float* __restrict__ bd0,
                                 const float* __restrict__ Wd1, const float* __restrict__ bd1,
                                 float* __restrict__ weff) {
    int tid = blockIdx.x * blockDim.x + threadIdx.x;
    if (tid < H_ + PM_) {
        float w = 0.f;
#pragma unroll 4
        for (int j = 0; j < 256; ++j) w += Wd1[j] * Wd0[(size_t)j * (H_ + PM_) + tid];
        weff[tid] = w;
    } else if (tid == H_ + PM_) {
        float bb = bd1[0];
        for (int j = 0; j < 256; ++j) bb += Wd1[j] * bd0[j];
        weff[H_ + PM_] = bb;
    }
}

__global__ __launch_bounds__(NTHREADS, 1)
void ealstm_kernel(const float* __restrict__ x,
                   const float* __restrict__ latlons,
                   const float* __restrict__ yearly,
                   const float* __restrict__ pred_month,
                   const float* __restrict__ Wf_i, const float* __restrict__ Wf_h, const float* __restrict__ bf_h,
                   const float* __restrict__ Wu,   const float* __restrict__ bu,
                   const float* __restrict__ Wg_i, const float* __restrict__ Wg_h, const float* __restrict__ bg_h,
                   const float* __restrict__ Wo_i, const float* __restrict__ Wo_h, const float* __restrict__ bo_h,
                   const float* __restrict__ weff,
                   float* __restrict__ out) {
    __shared__ float s_acc[3][BTILE][ACCP];                  // gate pre-activations (f32) ~50KB
    __shared__ float s_c[BTILE][H_];                         // cell state                  16KB
    __shared__ float s_ig[BTILE][H_];                        // static input gate           16KB
    __shared__ float s_bias[3][H_];                          //                              3KB
    __shared__ __align__(16) __bf16 s_h[BTILE][LDH];         // hidden state (A operand)     8KB
    __shared__ __align__(16) __bf16 s_x[BTILE][LDX];         // x_t slice   (A operand)      1KB
    __shared__ __align__(16) __bf16 s_wb[NWAVES][NFRAG * FRAG]; // Wo_h + Wi frags        176KB

    const int tid   = threadIdx.x;
    const int lane  = tid & 31;
    const int wave  = tid >> 5;
    const int bbase = blockIdx.x * BTILE;
    const int colbase = wave * 16;          // this wave's H-column tile

    // ---------------- setup ----------------
    for (int i = tid; i < H_; i += NTHREADS) {
        s_bias[0][i] = bf_h[i];
        s_bias[1][i] = bg_h[i];
        s_bias[2][i] = bo_h[i];
    }
    // h0 = c0 = 0; i_gate = sigmoid([latlons,yearly] @ Wu^T + bu)
    for (int e = tid; e < BTILE * H_; e += NTHREADS) {
        int r = e >> 8, h = e & (H_ - 1);
        s_c[r][h] = 0.f;
        s_h[r][h] = (__bf16)0.f;
        const float* wr = Wu + (size_t)h * S_;
        float acc = bu[h];
        acc += latlons[(size_t)(bbase + r) * 2 + 0] * wr[0];
        acc += latlons[(size_t)(bbase + r) * 2 + 1] * wr[1];
        const float* ya = yearly + (size_t)(bbase + r) * 37;
#pragma unroll 1
        for (int s = 0; s < 37; ++s) acc += ya[s] * wr[2 + s];
        s_ig[r][h] = fsig(acc);
    }

    // Persistent register fragments: f,g recurrent weights only (128 VGPRs).
    // o-gate recurrent + all input-projection frags go to LDS pre-swizzled.
    v16bf whF[8], whG[8];
#pragma unroll
    for (int k = 0; k < 8; ++k) {
        whF[k] = load_bfrag(Wf_h, H_, colbase, k * 32, lane);
        whG[k] = load_bfrag(Wg_h, H_, colbase, k * 32, lane);
        store_frag(&s_wb[wave][0], k, lane, load_bfrag(Wo_h, H_, colbase, k * 32, lane));
    }
    store_frag(&s_wb[wave][0],  8, lane, load_bfrag(Wf_i, D_, colbase, 0, lane));
    store_frag(&s_wb[wave][0],  9, lane, load_bfrag(Wg_i, D_, colbase, 0, lane));
    store_frag(&s_wb[wave][0], 10, lane, load_bfrag(Wo_i, D_, colbase, 0, lane));
    __syncthreads();

    const int nloc  = lane & 15;
    const int mbase = (lane < 16) ? 0 : 8;
    const float* xbase = x + (size_t)bbase * T_ * D_;

    // ---------------- recurrence over T ----------------
    for (int t = 0; t < T_; ++t) {
        // stage x[:, t, :] into LDS (bf16)
        if (tid < BTILE * D_) {
            int r = tid >> 5, d = tid & 31;
            const float* px = xbase + (size_t)r * T_ * D_ + (size_t)t * D_ + d;
            s_x[r][d] = (__bf16)(*px);
            if (t + 1 < T_) __builtin_prefetch(px + D_, 0, 1);  // global_prefetch next step
        }
        __syncthreads();

        // WMMA stage: acc = x_t @ Wi^T + h @ Wh^T for gates f,g,o of this wave's coltile
        {
            const __bf16* sw = &s_wb[wave][0];
            v8f aF = {}; v8f aG = {}; v8f aO = {};
            v16bf ax = load_afrag(&s_x[0][0], LDX, 0, lane);
            aF = wmma_bf16(ax, load_frag(sw,  8, lane), aF);
            aG = wmma_bf16(ax, load_frag(sw,  9, lane), aG);
            aO = wmma_bf16(ax, load_frag(sw, 10, lane), aO);
#pragma unroll
            for (int k = 0; k < 8; ++k) {
                v16bf ah = load_afrag(&s_h[0][0], LDH, k * 32, lane);
                v16bf wo = load_frag(sw, k, lane);
                aF = wmma_bf16(ah, whF[k], aF);
                aG = wmma_bf16(ah, whG[k], aG);
                aO = wmma_bf16(ah, wo, aO);
            }
#pragma unroll
            for (int r = 0; r < 8; ++r) {
                s_acc[0][mbase + r][colbase + nloc] = aF[r];
                s_acc[1][mbase + r][colbase + nloc] = aG[r];
                s_acc[2][mbase + r][colbase + nloc] = aO[r];
            }
        }
        __syncthreads();

        // combine stage: 256 threads, one H-column each
        if (tid < H_) {
            const int cc = tid;
            const float bf = s_bias[0][cc], bg = s_bias[1][cc], bo = s_bias[2][cc];
#pragma unroll
            for (int r = 0; r < BTILE; ++r) {
                float f  = fsig(s_acc[0][r][cc] + bf);
                float g  = ftanh(s_acc[1][r][cc] + bg);
                float o  = fsig(s_acc[2][r][cc] + bo);
                float cn = f * s_c[r][cc] + s_ig[r][cc] * g;
                s_c[r][cc] = cn;
                float hn = o * ftanh(cn);
                s_h[r][cc] = (__bf16)hn;
                s_acc[0][r][cc] = hn;   // keep f32 h for the head (valid after last step)
            }
        }
        __syncthreads();
    }

    // ---------------- folded dense head ----------------
    if (tid < BTILE) {
        const int r = tid;
        float sum = weff[H_ + PM_];
#pragma unroll 8
        for (int k = 0; k < H_; ++k) sum += s_acc[0][r][k] * weff[k];
        const float* pm = pred_month + (size_t)(bbase + r) * PM_;
#pragma unroll
        for (int j = 0; j < PM_; ++j) sum += pm[j] * weff[H_ + j];
        out[bbase + r] = sum;
    }
}

extern "C" void kernel_launch(void* const* d_in, const int* in_sizes, int n_in,
                              void* d_out, int out_size, void* d_ws, size_t ws_size,
                              hipStream_t stream) {
    const float* x       = (const float*)d_in[0];
    const float* latlons = (const float*)d_in[1];
    const float* yearly  = (const float*)d_in[2];
    const float* pmon    = (const float*)d_in[3];
    const float* Wf_i    = (const float*)d_in[4];
    const float* Wf_h    = (const float*)d_in[5];
    const float* bf_h    = (const float*)d_in[6];
    const float* Wu      = (const float*)d_in[7];
    const float* bu      = (const float*)d_in[8];
    const float* Wg_i    = (const float*)d_in[9];
    const float* Wg_h    = (const float*)d_in[10];
    const float* bg_h    = (const float*)d_in[11];
    const float* Wo_i    = (const float*)d_in[12];
    const float* Wo_h    = (const float*)d_in[13];
    const float* bo_h    = (const float*)d_in[14];
    const float* Wd0     = (const float*)d_in[15];
    const float* bd0     = (const float*)d_in[16];
    const float* Wd1     = (const float*)d_in[17];
    const float* bd1     = (const float*)d_in[18];
    float* out  = (float*)d_out;
    float* weff = (float*)d_ws;   // 269 floats

    head_fold_kernel<<<1, 512, 0, stream>>>(Wd0, bd0, Wd1, bd1, weff);

    ealstm_kernel<<<B_ / BTILE, NTHREADS, 0, stream>>>(
        x, latlons, yearly, pmon,
        Wf_i, Wf_h, bf_h, Wu, bu,
        Wg_i, Wg_h, bg_h, Wo_i, Wo_h, bo_h,
        weff, out);
}
